// IntentKG_80685255623083
// MI455X (gfx1250) — compile-verified, matrix operand order checked by reading, not run on hardware
//
#include <hip/hip_runtime.h>
#include <hip/hip_bf16.h>

typedef __attribute__((ext_vector_type(16))) _Float16 v16h;
typedef __attribute__((ext_vector_type(8)))  float    v8f;

// ---------------- problem constants ----------------
constexpr int cV0 = 1958;
constexpr int cV1 = 1430;
constexpr int cV2 = 131;
constexpr int cLEAF = cV0 + cV1 + cV2;      // 3519
constexpr int cENT  = 100000 - cLEAF;       // 96481
constexpr int cC    = 128;
constexpr int cEKG  = 500000;
constexpr int cNNZ  = 200000;
constexpr int cB = 16, cS = 8, cND = 40, cNP = 30, cNM = 30;
constexpr int cMED_PAD = cV2 + 2;           // 133
constexpr int cKEYPAD  = 3520;              // cLEAF padded to multiple of 32
constexpr int cSLD     = 40;                // padded LDS row stride (floats), 16B aligned reads

// ---------------- WMMA helpers (CDNA5 wave32) ----------------
__device__ __forceinline__ v8f wmma_f16(v16h a, v16h b, v8f c) {
  // D = A(16x32 f16) * B(32x16 f16) + C(16x16 f32)
  return __builtin_amdgcn_wmma_f32_16x16x32_f16(false, a, false, b, (short)0, c,
                                                false, false);
}

// A fragment (16x32): element j <-> K = kb + h*8 + j, element j+8 <-> K = kb+16+h*8+j
// row M = lane%16, h = lane/16  (ISA 16-bit A-matrix layout)
__device__ __forceinline__ v16h frag_a(const _Float16* p, int ld, int kb) {
  const int l = threadIdx.x & 31;
  const int m = l & 15, h = l >> 4;
  v16h a;
#pragma unroll
  for (int j = 0; j < 8; ++j) {
    a[j]     = p[m * ld + kb + h * 8 + j];
    a[j + 8] = p[m * ld + kb + 16 + h * 8 + j];
  }
  return a;
}

// B fragment (32x16), k-major storage: B[k][n] = p[(k0+k)*ld + n0+n]
__device__ __forceinline__ v16h frag_b_kmaj(const _Float16* p, int ld, int k0, int n0) {
  const int l = threadIdx.x & 31;
  const int n = l & 15, h = l >> 4;
  v16h b;
#pragma unroll
  for (int j = 0; j < 8; ++j) {
    b[j]     = p[(k0 + h * 8 + j) * ld + n0 + n];
    b[j + 8] = p[(k0 + 16 + h * 8 + j) * ld + n0 + n];
  }
  return b;
}

// B fragment (32x16), n-major storage (B = X^T): B[k][n] = p[(n0+n)*ld + k0+k]
__device__ __forceinline__ v16h frag_b_nmaj(const _Float16* p, int ld, int k0, int n0) {
  const int l = threadIdx.x & 31;
  const int n = l & 15, h = l >> 4;
  v16h b;
#pragma unroll
  for (int j = 0; j < 8; ++j) {
    b[j]     = p[(n0 + n) * ld + k0 + h * 8 + j];
    b[j + 8] = p[(n0 + n) * ld + k0 + 16 + h * 8 + j];
  }
  return b;
}

// B fragment from f32 source, k-major, converted inline to f16
__device__ __forceinline__ v16h frag_b_kmaj_f32(const float* p, int ld, int k0, int n0) {
  const int l = threadIdx.x & 31;
  const int n = l & 15, h = l >> 4;
  v16h b;
#pragma unroll
  for (int j = 0; j < 8; ++j) {
    b[j]     = (_Float16)p[(k0 + h * 8 + j) * ld + n0 + n];
    b[j + 8] = (_Float16)p[(k0 + 16 + h * 8 + j) * ld + n0 + n];
  }
  return b;
}

// ---------------- wave32 reduction ----------------
__device__ __forceinline__ float wave_sum(float v) {
#pragma unroll
  for (int m = 1; m < 32; m <<= 1) v += __shfl_xor(v, m, 32);
  return v;
}

// ---------------- graph-conv kernels ----------------
__global__ void k_deg(const int* __restrict__ eh, float* __restrict__ deg) {
  int i = blockIdx.x * blockDim.x + threadIdx.x;
  if (i < cEKG) atomicAdd(deg + eh[i], 1.0f);
}

__global__ void k_disen(const float* __restrict__ att, const float* __restrict__ kgw,
                        float* __restrict__ dw) {
  __shared__ float sm[8 * 49];
  const int t = threadIdx.x;   // 128 threads
  if (t < 8) {
    float mx = -1.0e30f;
    for (int j = 0; j < 49; ++j) mx = fmaxf(mx, att[t * 49 + j]);
    float s = 0.0f;
    for (int j = 0; j < 49; ++j) { float e = __expf(att[t * 49 + j] - mx); sm[t * 49 + j] = e; s += e; }
    const float inv = 1.0f / s;
    for (int j = 0; j < 49; ++j) sm[t * 49 + j] *= inv;
  }
  __syncthreads();
#pragma unroll
  for (int i = 0; i < 8; ++i) {
    float a = 0.0f;
    for (int j = 0; j < 49; ++j) a += sm[i * 49 + j] * kgw[j * cC + t];
    dw[i * cC + t] = a;
  }
}

// one wave per edge; lane owns 4 channels
__global__ void k_edge_scatter(const float* __restrict__ ent, const float* __restrict__ kgw,
                               const int* __restrict__ eh, const int* __restrict__ et,
                               const int* __restrict__ ety, float* __restrict__ agg) {
  const int wid = (blockIdx.x * blockDim.x + threadIdx.x) >> 5;
  if (wid >= cEKG) return;
  const int lane = threadIdx.x & 31;
  const int hn = eh[wid], tn = et[wid], r = ety[wid] - 1;
  const float4 tv = *(const float4*)(ent + (size_t)tn * cC + lane * 4);
  const float4 wv = *(const float4*)(kgw + (size_t)r  * cC + lane * 4);
  float* dst = agg + (size_t)hn * cC + lane * 4;
  atomicAdd(dst + 0, tv.x * wv.x);
  atomicAdd(dst + 1, tv.y * wv.y);
  atomicAdd(dst + 2, tv.z * wv.z);
  atomicAdd(dst + 3, tv.w * wv.w);
}

// one wave per nnz
__global__ void k_inter_scatter(const float* __restrict__ ent, const int* __restrict__ rows,
                                const int* __restrict__ cols, const float* __restrict__ vals,
                                float* __restrict__ uagg) {
  const int wid = (blockIdx.x * blockDim.x + threadIdx.x) >> 5;
  if (wid >= cNNZ) return;
  const int lane = threadIdx.x & 31;
  const int r = rows[wid], c = cols[wid];
  const float v = vals[wid];
  const float4 e = *(const float4*)(ent + (size_t)c * cC + lane * 4);
  float* dst = uagg + (size_t)r * cC + lane * 4;
  atomicAdd(dst + 0, v * e.x);
  atomicAdd(dst + 1, v * e.y);
  atomicAdd(dst + 2, v * e.z);
  atomicAdd(dst + 3, v * e.w);
}

// entity_emb = l2norm(ent_agg / max(deg,1))   (one wave per row, in-place)
__global__ void k_ent_update(const float* __restrict__ agg, const float* __restrict__ deg,
                             float* __restrict__ ent) {
  const int row = blockIdx.x * (blockDim.x >> 5) + (threadIdx.x >> 5);
  if (row >= cENT) return;
  const int lane = threadIdx.x & 31;
  const float d = fmaxf(deg[row], 1.0f);
  float4 v = *(const float4*)(agg + (size_t)row * cC + lane * 4);
  v.x /= d; v.y /= d; v.z /= d; v.w /= d;
  float ss = wave_sum(v.x * v.x + v.y * v.y + v.z * v.z + v.w * v.w);
  const float inv = 1.0f / fmaxf(sqrtf(ss), 1e-12f);
  v.x *= inv; v.y *= inv; v.z *= inv; v.w *= inv;
  *(float4*)(ent + (size_t)row * cC + lane * 4) = v;
}

// score softmax, usr_agg*(1+score@disen_w), l2norm, accumulate usr_res
__global__ void k_user_update(const float* __restrict__ uagg, float* __restrict__ uemb,
                              const float* __restrict__ latent, const float* __restrict__ dw,
                              float* __restrict__ ures) {
  const int row = blockIdx.x * (blockDim.x >> 5) + (threadIdx.x >> 5);
  if (row >= cLEAF) return;
  const int lane = threadIdx.x & 31;
  const float4 u = *(const float4*)(uemb + (size_t)row * cC + lane * 4);
  float sc[8];
#pragma unroll
  for (int i = 0; i < 8; ++i) {
    const float4 lt = *(const float4*)(latent + i * cC + lane * 4);
    sc[i] = wave_sum(u.x * lt.x + u.y * lt.y + u.z * lt.z + u.w * lt.w);
  }
  float mx = sc[0];
#pragma unroll
  for (int i = 1; i < 8; ++i) mx = fmaxf(mx, sc[i]);
  float se = 0.0f;
#pragma unroll
  for (int i = 0; i < 8; ++i) { sc[i] = __expf(sc[i] - mx); se += sc[i]; }
  const float sinv = 1.0f / se;
  float4 f = make_float4(0.f, 0.f, 0.f, 0.f);
#pragma unroll
  for (int i = 0; i < 8; ++i) {
    const float4 d = *(const float4*)(dw + i * cC + lane * 4);
    const float s = sc[i] * sinv;
    f.x += s * d.x; f.y += s * d.y; f.z += s * d.z; f.w += s * d.w;
  }
  const float4 a = *(const float4*)(uagg + (size_t)row * cC + lane * 4);
  float4 v;
  v.x = a.x * (1.0f + f.x); v.y = a.y * (1.0f + f.y);
  v.z = a.z * (1.0f + f.z); v.w = a.w * (1.0f + f.w);
  float ss = wave_sum(v.x * v.x + v.y * v.y + v.z * v.z + v.w * v.w);
  const float inv = 1.0f / fmaxf(sqrtf(ss), 1e-12f);
  v.x *= inv; v.y *= inv; v.z *= inv; v.w *= inv;
  *(float4*)(uemb + (size_t)row * cC + lane * 4) = v;
  float* rp = ures + (size_t)row * cC + lane * 4;
  rp[0] += v.x; rp[1] += v.y; rp[2] += v.z; rp[3] += v.w;
}

__global__ void k_leaf_f16(const float* __restrict__ src, _Float16* __restrict__ dst) {
  const int i = blockIdx.x * blockDim.x + threadIdx.x;
  if (i >= cKEYPAD * cC) return;
  const int row = i / cC;
  dst[i] = (row < cLEAF) ? (_Float16)src[i] : (_Float16)0.0f;
}

// ---------------- flash attention over leaf_intent (WMMA) ----------------
// One wave per 16-query tile. Q = table[idx] (16x128), keys/vals = leaf_intent f16.
// Online softmax stats are computed in A-layout (one row per lane; a single
// xor-16 shuffle completes each row reduction) so the P fragment is built
// directly in registers -- no serial butterfly ds_bpermute chains.
// Emits xout[q][0:128] = intent vec (0 if masked), xout[q][128:256] = ik vec.
__global__ __launch_bounds__(32) void k_flash(
    const int* __restrict__ idxs, const float* __restrict__ masks,
    const float* __restrict__ table, const _Float16* __restrict__ leafh,
    const float* __restrict__ leaff, _Float16* __restrict__ xout,
    int nq, int Nn, int shifted, int secBase, int secSize) {
  __shared__ _Float16 Qs[16 * cC];
  __shared__ float    Sf[16 * cSLD];        // raw logits staging (padded rows)
  __shared__ int   qidx[16];
  __shared__ float qmask[16];

  const int l  = threadIdx.x;
  const int qt = blockIdx.x * 16;

  if (l < 16) {
    int q = qt + l;
    int idx = 0; float mv = -1.0e9f;
    if (q < nq) {
      const int b = q / (cS * Nn), s = (q / Nn) % cS, n = q % Nn;
      if (shifted) {
        if (s == 0) { idx = cMED_PAD; mv = -1.0e9f; }
        else        { idx = idxs[(b * cS + s - 1) * Nn + n]; mv = masks[(b * cS + s - 1) * Nn + n]; }
      } else { idx = idxs[q]; mv = masks[q]; }
    }
    qidx[l] = idx; qmask[l] = mv;
  }
  __syncthreads();

  // build Q tile (f32 table gather -> f16 LDS)
  for (int t = l; t < 16 * 32; t += 32) {
    const int m = t >> 5, c4 = (t & 31) * 4;
    const float* src = table + (size_t)qidx[m] * cC + c4;
#pragma unroll
    for (int k = 0; k < 4; ++k) Qs[m * cC + c4 + k] = (_Float16)src[k];
  }
  __syncthreads();

  v16h qa[4];
#pragma unroll
  for (int kk = 0; kk < 4; ++kk) qa[kk] = frag_a(Qs, cC, kk * 32);

  v8f o[8];
#pragma unroll
  for (int f = 0; f < 8; ++f)
#pragma unroll
    for (int i = 0; i < 8; ++i) o[f][i] = 0.0f;

  const int col  = l & 15;   // D-layout column
  const int h    = l >> 4;   // lane half
  const int qrow = l & 15;   // A-layout row owned by this lane

  // per-lane online-softmax state for row `qrow` (duplicated in both halves)
  float mrow = -1.0e30f, lrow = 0.0f;

  for (int kt = 0; kt < cKEYPAD; kt += 32) {
    if (kt + 32 < cKEYPAD)
      __builtin_prefetch((const void*)(leafh + (size_t)(kt + 32) * cC), 0, 0);

    // ---- S = Q @ K^T for 32 keys (two 16x16 D tiles) ----
    v8f s0, s1;
#pragma unroll
    for (int i = 0; i < 8; ++i) { s0[i] = 0.0f; s1[i] = 0.0f; }
#pragma unroll
    for (int kk = 0; kk < 4; ++kk) {
      s0 = wmma_f16(qa[kk], frag_b_nmaj(leafh, cC, kk * 32, kt),      s0);
      s1 = wmma_f16(qa[kk], frag_b_nmaj(leafh, cC, kk * 32, kt + 16), s1);
    }

    // ---- stage raw logits to LDS (D-layout scatter) ----
    __syncthreads();                       // previous iteration's Sf reads done
#pragma unroll
    for (int r = 0; r < 8; ++r) {
      Sf[(r + 8 * h) * cSLD + col]      = s0[r];
      Sf[(r + 8 * h) * cSLD + 16 + col] = s1[r];
    }
    __syncthreads();

    // ---- A-layout softmax: lane owns 16 logits of row `qrow` in frag_a k order ----
    float e[16];
    float lmax = -1.0e30f;
#pragma unroll
    for (int j = 0; j < 16; ++j) {
      const int k = (j < 8) ? (h * 8 + j) : (16 + h * 8 + (j - 8));  // frag_a element order
      float x = Sf[qrow * cSLD + k];
      if (kt + k >= cLEAF) x = -1.0e30f;   // padded keys
      e[j] = x;
      lmax = fmaxf(lmax, x);
    }
    lmax = fmaxf(lmax, __shfl_xor(lmax, 16, 32));    // other half of the row
    const float mn = fmaxf(mrow, lmax);
    float lsum = 0.0f;
#pragma unroll
    for (int j = 0; j < 16; ++j) { e[j] = __expf(e[j] - mn); lsum += e[j]; }
    lsum += __shfl_xor(lsum, 16, 32);
    const float scl = __expf(mrow - mn);
    lrow = lrow * scl + lsum;
    mrow = mn;

    // P fragment directly in registers (already A-layout)
    v16h pa;
#pragma unroll
    for (int j = 0; j < 16; ++j) pa[j] = (_Float16)e[j];

    // broadcast row scales to this lane's D-layout rows (r + 8h)
    float oscale[8];
#pragma unroll
    for (int r = 0; r < 8; ++r) oscale[r] = __shfl(scl, r + 8 * h, 32);
#pragma unroll
    for (int f = 0; f < 8; ++f)
#pragma unroll
      for (int r = 0; r < 8; ++r) o[f][r] *= oscale[r];

    // ---- O += P @ V ----
#pragma unroll
    for (int f = 0; f < 8; ++f)
      o[f] = wmma_f16(pa, frag_b_kmaj(leafh, cC, kt, f * 16), o[f]);
  }

  // gather per-row normalizers into D-layout
  float lfin[8];
#pragma unroll
  for (int r = 0; r < 8; ++r) lfin[r] = __shfl(lrow, r + 8 * h, 32);

  // finalize intent half of x
#pragma unroll
  for (int f = 0; f < 8; ++f)
#pragma unroll
    for (int r = 0; r < 8; ++r) {
      const int rr = r + 8 * h, q = qt + rr;
      if (q < nq) {
        const float val = o[f][r] / fmaxf(lfin[r], 1e-30f);
        const bool dead = (qmask[rr] != 0.0f);
        xout[(size_t)q * 256 + f * 16 + col] = dead ? (_Float16)0.0f : (_Float16)val;
      }
    }

  // ik half of x (never masked; out-of-section index -> 0)
  for (int t = l; t < 16 * 32; t += 32) {
    const int m = t >> 5, c4 = (t & 31) * 4;
    const int q = qt + m, idx = qidx[m];
    const bool ok = (q < nq) && (idx < secSize);
    const float* src = leaff + (size_t)(secBase + (ok ? idx : 0)) * cC + c4;
#pragma unroll
    for (int k = 0; k < 4; ++k)
      xout[(size_t)q * 256 + cC + c4 + k] = ok ? (_Float16)src[k] : (_Float16)0.0f;
  }
}

// ---------------- tanh([int|ik] @ W + b), sum over n (WMMA + atomics) --------
__global__ __launch_bounds__(32) void k_agg(
    const _Float16* __restrict__ x, const float* __restrict__ w,
    const float* __restrict__ bias, float* __restrict__ curt, int nq, int Nn) {
  const int mt = blockIdx.x * 16;
  const int nt = blockIdx.y * 16;
  v8f acc;
#pragma unroll
  for (int i = 0; i < 8; ++i) acc[i] = 0.0f;
#pragma unroll
  for (int kk = 0; kk < 256; kk += 32) {
    const v16h a = frag_a(x + (size_t)mt * 256, 256, kk);
    const v16h b = frag_b_kmaj_f32(w, cC, kk, nt);
    acc = wmma_f16(a, b, acc);
  }
  const int l = threadIdx.x, col = nt + (l & 15), h = l >> 4;
  const float bb = bias[col];
#pragma unroll
  for (int r = 0; r < 8; ++r) {
    const int q = mt + r + 8 * h;
    if (q < nq) atomicAdd(curt + (size_t)(q / Nn) * cC + col, tanhf(acc[r] + bb));
  }
}

// ---------------- final visit GEMM ----------------
__device__ __forceinline__ float cat3(const float* a, const float* b, const float* c,
                                      int row, int k) {
  if (k < cC)     return a[row * cC + k];
  if (k < 2 * cC) return b[row * cC + k - cC];
  return c[row * cC + k - 2 * cC];
}

__global__ __launch_bounds__(32) void k_final(
    const float* __restrict__ cd, const float* __restrict__ cp,
    const float* __restrict__ cm, const float* __restrict__ w,
    const float* __restrict__ bias, float* __restrict__ out) {
  const int mt = (blockIdx.x >> 3) * 16;
  const int nt = (blockIdx.x & 7) * 16;
  const int l = threadIdx.x, m = l & 15, h = l >> 4, col = l & 15;
  v8f acc;
#pragma unroll
  for (int i = 0; i < 8; ++i) acc[i] = 0.0f;
#pragma unroll
  for (int kk = 0; kk < 384; kk += 32) {
    v16h a;
#pragma unroll
    for (int j = 0; j < 8; ++j) {
      a[j]     = (_Float16)cat3(cd, cp, cm, mt + m, kk + h * 8 + j);
      a[j + 8] = (_Float16)cat3(cd, cp, cm, mt + m, kk + 16 + h * 8 + j);
    }
    const v16h b = frag_b_kmaj_f32(w, cC, kk, nt);
    acc = wmma_f16(a, b, acc);
  }
  const float bb = bias[nt + col];
#pragma unroll
  for (int r = 0; r < 8; ++r)
    out[(size_t)(mt + r + 8 * h) * cC + nt + col] = tanhf(acc[r] + bb);
}

// ---------------- host launcher ----------------
extern "C" void kernel_launch(void* const* d_in, const int* in_sizes, int n_in,
                              void* d_out, int out_size, void* d_ws, size_t ws_size,
                              hipStream_t stream) {
  (void)in_sizes; (void)n_in; (void)out_size; (void)ws_size;
  const int*   diseases    = (const int*)d_in[0];
  const int*   procedures  = (const int*)d_in[1];
  const int*   medications = (const int*)d_in[2];
  const float* d_mask      = (const float*)d_in[3];
  const float* p_mask      = (const float*)d_in[4];
  const float* m_mask      = (const float*)d_in[5];
  const int*   edge_head   = (const int*)d_in[6];
  const int*   edge_tail   = (const int*)d_in[7];
  const int*   edge_type   = (const int*)d_in[8];
  const int*   inter_rows  = (const int*)d_in[9];
  const int*   inter_cols  = (const int*)d_in[10];
  const float* inter_vals  = (const float*)d_in[11];
  const float* all_embed   = (const float*)d_in[12];
  const float* latent_emb  = (const float*)d_in[13];
  const float* kg_weight   = (const float*)d_in[14];
  const float* disen_att   = (const float*)d_in[15];
  const float* diag_table  = (const float*)d_in[16];
  const float* proc_table  = (const float*)d_in[17];
  const float* med_table   = (const float*)d_in[18];
  const float* diag_w      = (const float*)d_in[19];
  const float* diag_b      = (const float*)d_in[20];
  const float* proc_w      = (const float*)d_in[21];
  const float* proc_b      = (const float*)d_in[22];
  const float* med_w       = (const float*)d_in[23];
  const float* med_b       = (const float*)d_in[24];
  const float* curt_w      = (const float*)d_in[25];
  const float* curt_b      = (const float*)d_in[26];
  float* out = (float*)d_out;

  char* wp = (char*)d_ws;
  auto carve = [&](size_t bytes) -> void* {
    void* p = (void*)wp;
    wp += (bytes + 255) & ~(size_t)255;
    return p;
  };
  float*    ent   = (float*)carve((size_t)cENT  * cC * 4);
  float*    eagg  = (float*)carve((size_t)cENT  * cC * 4);
  float*    uemb  = (float*)carve((size_t)cLEAF * cC * 4);
  float*    uagg  = (float*)carve((size_t)cLEAF * cC * 4);
  float*    ures  = (float*)carve((size_t)cLEAF * cC * 4);
  float*    deg   = (float*)carve((size_t)cENT * 4);
  float*    dw    = (float*)carve((size_t)8 * cC * 4);
  _Float16* leafh = (_Float16*)carve((size_t)cKEYPAD * cC * 2);
  _Float16* xbuf  = (_Float16*)carve((size_t)(cB * cS * cND) * 256 * 2);
  float*    curtd = (float*)carve((size_t)cB * cS * cC * 4);
  float*    curtp = (float*)carve((size_t)cB * cS * cC * 4);
  float*    curtm = (float*)carve((size_t)cB * cS * cC * 4);

  // split all_embed; usr_res starts as user_emb
  hipMemcpyAsync(uemb, all_embed, (size_t)cLEAF * cC * 4, hipMemcpyDeviceToDevice, stream);
  hipMemcpyAsync(ures, all_embed, (size_t)cLEAF * cC * 4, hipMemcpyDeviceToDevice, stream);
  hipMemcpyAsync(ent, all_embed + (size_t)cLEAF * cC, (size_t)cENT * cC * 4,
                 hipMemcpyDeviceToDevice, stream);
  hipMemsetAsync(deg, 0, (size_t)cENT * 4, stream);
  k_deg<<<(cEKG + 255) / 256, 256, 0, stream>>>(edge_head, deg);
  k_disen<<<1, 128, 0, stream>>>(disen_att, kg_weight, dw);

  for (int hop = 0; hop < 3; ++hop) {
    hipMemsetAsync(eagg, 0, (size_t)cENT  * cC * 4, stream);
    hipMemsetAsync(uagg, 0, (size_t)cLEAF * cC * 4, stream);
    k_edge_scatter<<<(cEKG * 32 + 255) / 256, 256, 0, stream>>>(
        ent, kg_weight, edge_head, edge_tail, edge_type, eagg);
    k_inter_scatter<<<(cNNZ * 32 + 255) / 256, 256, 0, stream>>>(
        ent, inter_rows, inter_cols, inter_vals, uagg);
    k_user_update<<<(cLEAF + 7) / 8, 256, 0, stream>>>(uagg, uemb, latent_emb, dw, ures);
    k_ent_update<<<(cENT + 7) / 8, 256, 0, stream>>>(eagg, deg, ent);
  }
  k_leaf_f16<<<(cKEYPAD * cC + 255) / 256, 256, 0, stream>>>(ures, leafh);

  hipMemsetAsync(curtd, 0, (size_t)cB * cS * cC * 4, stream);
  hipMemsetAsync(curtp, 0, (size_t)cB * cS * cC * 4, stream);
  hipMemsetAsync(curtm, 0, (size_t)cB * cS * cC * 4, stream);

  const int nqd = cB * cS * cND, nqp = cB * cS * cNP, nqm = cB * cS * cNM;

  k_flash<<<nqd / 16, 32, 0, stream>>>(diseases, d_mask, diag_table, leafh, ures, xbuf,
                                       nqd, cND, 0, 0, cV0);
  k_agg<<<dim3(nqd / 16, 8), 32, 0, stream>>>(xbuf, diag_w, diag_b, curtd, nqd, cND);

  k_flash<<<nqp / 16, 32, 0, stream>>>(procedures, p_mask, proc_table, leafh, ures, xbuf,
                                       nqp, cNP, 0, cV0, cV1);
  k_agg<<<dim3(nqp / 16, 8), 32, 0, stream>>>(xbuf, proc_w, proc_b, curtp, nqp, cNP);

  k_flash<<<nqm / 16, 32, 0, stream>>>(medications, m_mask, med_table, leafh, ures, xbuf,
                                       nqm, cNM, 1, cV0 + cV1, cV2);
  k_agg<<<dim3(nqm / 16, 8), 32, 0, stream>>>(xbuf, med_w, med_b, curtm, nqm, cNM);

  k_final<<<64, 32, 0, stream>>>(curtd, curtp, curtm, curt_w, curt_b, out);
}
